// CS2TransformerEncoderLayer_14396730376973
// MI455X (gfx1250) — matrix operand order repaired and checked by reading, compiled, not run on hardware
//
#include <hip/hip_runtime.h>
#include <hip/hip_bf16.h>
#include <math.h>

typedef __bf16 bf16_t;
typedef __attribute__((ext_vector_type(16))) __bf16 v16bf;
typedef __attribute__((ext_vector_type(8)))  float  v8f;

#define S_LEN  4096
#define BATCH  2
#define DMODEL 512
#define NHEAD  8
#define DHEAD  64
#define FFN    2048
#define MROWS  (S_LEN*BATCH)   // 8192

union FragB { uint4 u[2]; v16bf v; };

// Low 32 bits of a generic pointer into __shared__ are the hardware LDS byte
// address (generic->LDS mapping truncates to addr[31:0], see ISA 10.2).
__device__ __forceinline__ uint32_t lds_addr(const void* p) {
    return (uint32_t)(uintptr_t)p;
}
// Async DMA: 16 bytes global -> LDS per lane, tracked by ASYNCcnt.
__device__ __forceinline__ void async_copy_b128(uint32_t lds, const bf16_t* g) {
    asm volatile("global_load_async_to_lds_b128 %0, %1, off"
                 :: "v"(lds), "v"(g) : "memory");
}
__device__ __forceinline__ void wait_async0() {
    asm volatile("s_wait_asynccnt 0x0" ::: "memory");
}

// ---------------------------------------------------------------- convert
__global__ void cvt_f32_to_bf16(const float* __restrict__ in, bf16_t* __restrict__ out, int n) {
    int i = blockIdx.x * blockDim.x + threadIdx.x;
    if (i < n) out[i] = (bf16_t)in[i];
}

// ---------------------------------------------------------------- LayerNorm (+optional RoPE)
// One block (256 threads) per row of [MROWS, 512].
template<bool DO_ROPE>
__global__ void ln_kernel(const float* __restrict__ x,
                          const float* __restrict__ w, const float* __restrict__ bvec,
                          const float* __restrict__ rcos, const float* __restrict__ rsin,
                          bf16_t* __restrict__ out_ln, bf16_t* __restrict__ out_rope) {
    const int row = blockIdx.x;          // row = s*BATCH + b
    const int t   = threadIdx.x;         // 0..255
    __shared__ float xs[DMODEL];
    __shared__ float red[256];

    float a0 = x[(size_t)row * DMODEL + t];
    float a1 = x[(size_t)row * DMODEL + t + 256];
    red[t] = a0 + a1;
    __syncthreads();
    for (int off = 128; off > 0; off >>= 1) { if (t < off) red[t] += red[t + off]; __syncthreads(); }
    float mean = red[0] * (1.0f / DMODEL);
    __syncthreads();
    float d0 = a0 - mean, d1 = a1 - mean;
    red[t] = d0 * d0 + d1 * d1;
    __syncthreads();
    for (int off = 128; off > 0; off >>= 1) { if (t < off) red[t] += red[t + off]; __syncthreads(); }
    float rstd = rsqrtf(red[0] * (1.0f / DMODEL) + 1e-5f);

    float y0 = d0 * rstd * w[t]       + bvec[t];
    float y1 = d1 * rstd * w[t + 256] + bvec[t + 256];
    out_ln[(size_t)row * DMODEL + t]       = (bf16_t)y0;
    out_ln[(size_t)row * DMODEL + t + 256] = (bf16_t)y1;

    if (DO_ROPE) {
        xs[t] = y0; xs[t + 256] = y1;
        __syncthreads();
        const int s = row / BATCH;
        #pragma unroll
        for (int half = 0; half < 2; half++) {
            int col = t + half * 256;
            int h   = col >> 6;          // head
            int dh  = col & 63;          // within head
            float c  = rcos[s * DHEAD + dh];
            float sn = rsin[s * DHEAD + dh];
            // rotate_half: cat(-x[1::2], x[::2])
            float r = (dh < 32) ? -xs[(h << 6) + 2 * dh + 1]
                                :  xs[(h << 6) + 2 * (dh - 32)];
            out_rope[(size_t)row * DMODEL + col] = (bf16_t)(xs[col] * c + r * sn);
        }
    }
}

// ---------------------------------------------------------------- WMMA GEMM
// C[M,N] = epilogue( A[M,K](bf16) * W[N,K]^T(bf16) + bias )
// Block: 256 threads = 8 waves in 4(M) x 2(N); block tile 128x128; K-step 32.
// Double-buffered LDS filled by global_load_async_to_lds_b128 (ASYNCcnt path):
// next tile's DMA overlaps current tile's WMMAs; one barrier per K-step.
#define EPI_BF16 0
#define EPI_GELU 1
#define EPI_RES  2

template<int EPI>
__global__ __launch_bounds__(256)
void gemm_bf16(const bf16_t* __restrict__ A, const bf16_t* __restrict__ W,
               const float* __restrict__ bias, const float* __restrict__ resid,
               bf16_t* __restrict__ Cb, float* __restrict__ Cf,
               int M, int N, int K, int lda) {
    __shared__ bf16_t As[2][128 * 32];
    __shared__ bf16_t Ws[2][128 * 32];
    const int tid   = threadIdx.x;
    const int lane  = tid & 31;
    const int wv    = tid >> 5;
    const int wm    = wv >> 1;          // 0..3
    const int wn    = wv & 1;           // 0..1
    const int l16   = lane & 15;
    const int lhalf = lane >> 4;
    const int mB = blockIdx.y * 128;
    const int nB = blockIdx.x * 128;

    v8f zero8 = {0.f,0.f,0.f,0.f,0.f,0.f,0.f,0.f};
    v8f acc[2][4];
    #pragma unroll
    for (int ms = 0; ms < 2; ms++)
        #pragma unroll
        for (int ns = 0; ns < 4; ns++) acc[ms][ns] = zero8;

    // staging pattern: thread covers rows lr, lr+64 at k-offset lp (8 bf16 = 16B)
    const int lr = tid >> 2;      // 0..63
    const int lp = (tid & 3) * 8; // 0/8/16/24
    const bf16_t* gA0 = A + (size_t)(mB + lr) * lda + lp;
    const bf16_t* gA1 = gA0 + (size_t)64 * lda;
    const bf16_t* gW0 = W + (size_t)(nB + lr) * K + lp;
    const bf16_t* gW1 = gW0 + (size_t)64 * K;
    uint32_t sA0[2], sA1[2], sW0[2], sW1[2];
    #pragma unroll
    for (int bi = 0; bi < 2; bi++) {
        sA0[bi] = lds_addr(&As[bi][lr * 32 + lp]);
        sA1[bi] = lds_addr(&As[bi][(lr + 64) * 32 + lp]);
        sW0[bi] = lds_addr(&Ws[bi][lr * 32 + lp]);
        sW1[bi] = lds_addr(&Ws[bi][(lr + 64) * 32 + lp]);
    }

    const int KT = K / 32;
    // prologue: fill buffer 0
    async_copy_b128(sA0[0], gA0);
    async_copy_b128(sA1[0], gA1);
    async_copy_b128(sW0[0], gW0);
    async_copy_b128(sW1[0], gW1);

    for (int kt = 0; kt < KT; kt++) {
        wait_async0();
        __syncthreads();               // buffer kt&1 visible to all waves
        if (kt + 1 < KT) {             // prefetch next tile into other buffer
            const int nb = (kt + 1) & 1;
            const int ko = (kt + 1) * 32;
            async_copy_b128(sA0[nb], gA0 + ko);
            async_copy_b128(sA1[nb], gA1 + ko);
            async_copy_b128(sW0[nb], gW0 + ko);
            async_copy_b128(sW1[nb], gW1 + ko);
        }
        const bf16_t* as = As[kt & 1];
        const bf16_t* wsb = Ws[kt & 1];

        FragB afr[2], bfr[4];
        const int koffA = lhalf * 8;    // A frag: K chunks {koffA..+8, koffA+16..+24}
        #pragma unroll
        for (int ms = 0; ms < 2; ms++) {
            int arow = wm * 32 + ms * 16 + l16;
            afr[ms].u[0] = *reinterpret_cast<const uint4*>(&as[arow * 32 + koffA]);
            afr[ms].u[1] = *reinterpret_cast<const uint4*>(&as[arow * 32 + koffA + 16]);
        }
        const int koffB = lhalf * 8;    // B frag: lane%16 = N col, K chunks {koffB..+8, koffB+16..+24}
        #pragma unroll
        for (int ns = 0; ns < 4; ns++) {
            int nrow = wn * 64 + ns * 16 + l16;
            bfr[ns].u[0] = *reinterpret_cast<const uint4*>(&wsb[nrow * 32 + koffB]);
            bfr[ns].u[1] = *reinterpret_cast<const uint4*>(&wsb[nrow * 32 + koffB + 16]);
        }
        #pragma unroll
        for (int ms = 0; ms < 2; ms++)
            #pragma unroll
            for (int ns = 0; ns < 4; ns++)
                acc[ms][ns] = __builtin_amdgcn_wmma_f32_16x16x32_bf16(
                    false, afr[ms].v, false, bfr[ns].v, (short)0, acc[ms][ns], false, false);
        // no trailing barrier: next iteration's wait_async0 + barrier protects the
        // buffer swap, and the prefetch always targets the buffer all waves
        // finished reading before this iteration's barrier.
    }

    // epilogue: C layout -> row = base + e + 8*lhalf, col = base + lane%16
    #pragma unroll
    for (int ms = 0; ms < 2; ms++) {
        int rowb = mB + wm * 32 + ms * 16 + 8 * lhalf;
        #pragma unroll
        for (int ns = 0; ns < 4; ns++) {
            int col = nB + wn * 64 + ns * 16 + l16;
            float bsv = bias[col];
            #pragma unroll
            for (int e = 0; e < 8; e++) {
                int rr = rowb + e;
                float v = acc[ms][ns][e] + bsv;
                if (EPI == EPI_GELU) v = 0.5f * v * (1.0f + erff(v * 0.70710678118f));
                if (EPI == EPI_RES)  Cf[(size_t)rr * N + col] = v + resid[(size_t)rr * N + col];
                else                 Cb[(size_t)rr * N + col] = (bf16_t)v;
            }
        }
    }
}

// ---------------------------------------------------------------- flash attention (causal)
// Grid: (S/64 q-blocks, B*H). Block: 128 threads = 4 waves, each wave owns 16 q rows.
__global__ __launch_bounds__(128)
void attn_kernel(const bf16_t* __restrict__ qk,   // [MROWS, 1024] : Q | K per head
                 const bf16_t* __restrict__ vbuf, // [MROWS, 512]
                 bf16_t* __restrict__ attn) {     // [MROWS, 512]
    const int bx   = blockIdx.x;           // q-tile 0..63
    const int bh   = blockIdx.y;           // 0..15
    const int b    = bh / NHEAD, h = bh % NHEAD;
    const int tid  = threadIdx.x;
    const int lane = tid & 31, wv = tid >> 5;
    const int l16  = lane & 15, lhalf = lane >> 4;

    __shared__ bf16_t Vt[64 * 72];         // V transposed: [d][key], padded to 72
    __shared__ bf16_t Pw[4][16 * 72];      // per-wave P tile: [qrow][key], padded

    const int qb0 = bx * 64;
    const int q0w = qb0 + wv * 16;

    // Hoist Q fragments (A layout; K-dim = Dh = 64 -> two 32-chunks)
    FragB qf[2];
    {
        const int qrow = q0w + l16;
        const bf16_t* qp = qk + ((size_t)qrow * BATCH + b) * 1024 + h * 64;
        const int koffA = lhalf * 8;
        #pragma unroll
        for (int kk = 0; kk < 2; kk++) {
            qf[kk].u[0] = *reinterpret_cast<const uint4*>(qp + kk * 32 + koffA);
            qf[kk].u[1] = *reinterpret_cast<const uint4*>(qp + kk * 32 + koffA + 16);
        }
    }

    v8f zero8 = {0.f,0.f,0.f,0.f,0.f,0.f,0.f,0.f};
    v8f o[4];
    #pragma unroll
    for (int nd = 0; nd < 4; nd++) o[nd] = zero8;
    float m_run[8], l_run[8];
    #pragma unroll
    for (int e = 0; e < 8; e++) { m_run[e] = -3.0e38f; l_run[e] = 0.0f; }

    const int ktiles = bx + 1;             // causal: keys 0 .. qb0+63
    for (int kt = 0; kt < ktiles; kt++) {
        const int kb = kt * 64;

        // stage V tile transposed into LDS: Vt[d][key]
        {
            const int key  = tid >> 1;
            const int dh0  = (tid & 1) * 32;
            const bf16_t* vp = vbuf + ((size_t)(kb + key) * BATCH + b) * DMODEL + h * 64 + dh0;
            union { uint4 u[4]; bf16_t hh[32]; } vl;
            vl.u[0] = *reinterpret_cast<const uint4*>(vp);
            vl.u[1] = *reinterpret_cast<const uint4*>(vp + 8);
            vl.u[2] = *reinterpret_cast<const uint4*>(vp + 16);
            vl.u[3] = *reinterpret_cast<const uint4*>(vp + 24);
            #pragma unroll
            for (int i = 0; i < 32; i++) Vt[(dh0 + i) * 72 + key] = vl.hh[i];
        }
        __syncthreads();

        // scores S = Q K^T  (K frags straight from global: contiguous along Dh)
        v8f sc[4];
        #pragma unroll
        for (int ns = 0; ns < 4; ns++) {
            v8f a = zero8;
            #pragma unroll
            for (int kk = 0; kk < 2; kk++) {
                FragB kf;
                int krow = kb + ns * 16 + l16;
                const bf16_t* kp = qk + ((size_t)krow * BATCH + b) * 1024 + 512 + h * 64
                                      + kk * 32 + lhalf * 8;
                kf.u[0] = *reinterpret_cast<const uint4*>(kp);
                kf.u[1] = *reinterpret_cast<const uint4*>(kp + 16);
                a = __builtin_amdgcn_wmma_f32_16x16x32_bf16(
                        false, qf[kk].v, false, kf.v, (short)0, a, false, false);
            }
            sc[ns] = a;
        }

        // scale + causal mask
        #pragma unroll
        for (int ns = 0; ns < 4; ns++) {
            int key = kb + ns * 16 + l16;
            #pragma unroll
            for (int e = 0; e < 8; e++) {
                int qr = q0w + e + 8 * lhalf;
                float v = sc[ns][e] * 0.125f;   // 1/sqrt(64)
                sc[ns][e] = (key > qr) ? -3.0e38f : v;
            }
        }

        // online softmax (row lives in one 16-lane half-wave)
        float alpha[8];
        #pragma unroll
        for (int e = 0; e < 8; e++) {
            float tmax = fmaxf(fmaxf(sc[0][e], sc[1][e]), fmaxf(sc[2][e], sc[3][e]));
            #pragma unroll
            for (int mk = 8; mk >= 1; mk >>= 1) tmax = fmaxf(tmax, __shfl_xor(tmax, mk, 32));
            float mnew = fmaxf(m_run[e], tmax);
            alpha[e] = __expf(m_run[e] - mnew);
            m_run[e] = mnew;
            float rs = 0.f;
            #pragma unroll
            for (int ns = 0; ns < 4; ns++) { float p = __expf(sc[ns][e] - mnew); sc[ns][e] = p; rs += p; }
            #pragma unroll
            for (int mk = 8; mk >= 1; mk >>= 1) rs += __shfl_xor(rs, mk, 32);
            l_run[e] = l_run[e] * alpha[e] + rs;
        }
        #pragma unroll
        for (int nd = 0; nd < 4; nd++)
            #pragma unroll
            for (int e = 0; e < 8; e++) o[nd][e] *= alpha[e];

        // P: C layout -> A layout via LDS round trip
        #pragma unroll
        for (int ns = 0; ns < 4; ns++)
            #pragma unroll
            for (int e = 0; e < 8; e++)
                Pw[wv][(e + 8 * lhalf) * 72 + ns * 16 + l16] = (bf16_t)sc[ns][e];
        __syncthreads();

        // O += P V  (K-dim = 64 keys -> two 32-chunks)
        #pragma unroll
        for (int kk = 0; kk < 2; kk++) {
            FragB pf;
            const int koffA = lhalf * 8;
            pf.u[0] = *reinterpret_cast<uint4*>(&Pw[wv][l16 * 72 + kk * 32 + koffA]);
            pf.u[1] = *reinterpret_cast<uint4*>(&Pw[wv][l16 * 72 + kk * 32 + koffA + 16]);
            #pragma unroll
            for (int nd = 0; nd < 4; nd++) {
                FragB vf;
                int dcol = nd * 16 + l16;
                vf.u[0] = *reinterpret_cast<uint4*>(&Vt[dcol * 72 + kk * 32 + lhalf * 8]);
                vf.u[1] = *reinterpret_cast<uint4*>(&Vt[dcol * 72 + kk * 32 + lhalf * 8 + 16]);
                o[nd] = __builtin_amdgcn_wmma_f32_16x16x32_bf16(
                            false, pf.v, false, vf.v, (short)0, o[nd], false, false);
            }
        }
        __syncthreads();
    }

    // normalize and store
    #pragma unroll
    for (int nd = 0; nd < 4; nd++) {
        #pragma unroll
        for (int e = 0; e < 8; e++) {
            int qr = q0w + e + 8 * lhalf;
            float val = o[nd][e] / l_run[e];
            attn[((size_t)qr * BATCH + b) * DMODEL + h * 64 + nd * 16 + l16] = (bf16_t)val;
        }
    }
}

// ---------------------------------------------------------------- launch
extern "C" void kernel_launch(void* const* d_in, const int* in_sizes, int n_in,
                              void* d_out, int out_size, void* d_ws, size_t ws_size,
                              hipStream_t stream) {
    const float* src    = (const float*)d_in[0];
    const float* rcos   = (const float*)d_in[1];
    const float* rsin   = (const float*)d_in[2];
    const float* inpw   = (const float*)d_in[3];
    const float* inpb   = (const float*)d_in[4];
    const float* outw   = (const float*)d_in[5];
    const float* outb   = (const float*)d_in[6];
    const float* w1     = (const float*)d_in[7];
    const float* b1     = (const float*)d_in[8];
    const float* w2     = (const float*)d_in[9];
    const float* b2     = (const float*)d_in[10];
    const float* ln1w   = (const float*)d_in[11];
    const float* ln1b   = (const float*)d_in[12];
    const float* ln2w   = (const float*)d_in[13];
    const float* ln2b   = (const float*)d_in[14];
    float* outp = (float*)d_out;

    // workspace carve-up (256B aligned)
    char* ws = (char*)d_ws;
    size_t off = 0;
    auto carve = [&](size_t bytes) { char* p = ws + off; off = (off + bytes + 255) & ~(size_t)255; return p; };
    bf16_t* w_in_bf  = (bf16_t*)carve((size_t)1536 * 512 * 2);
    bf16_t* w_out_bf = (bf16_t*)carve((size_t)512 * 512 * 2);
    bf16_t* w1_bf    = (bf16_t*)carve((size_t)2048 * 512 * 2);
    bf16_t* w2_bf    = (bf16_t*)carve((size_t)512 * 2048 * 2);
    bf16_t* xln      = (bf16_t*)carve((size_t)MROWS * 512 * 2);
    bf16_t* xr       = (bf16_t*)carve((size_t)MROWS * 512 * 2);
    bf16_t* qkb      = (bf16_t*)carve((size_t)MROWS * 1024 * 2);
    bf16_t* vb       = (bf16_t*)carve((size_t)MROWS * 512 * 2);
    bf16_t* attnb    = (bf16_t*)carve((size_t)MROWS * 512 * 2);
    float*  res1     = (float*) carve((size_t)MROWS * 512 * 4);
    bf16_t* yb       = (bf16_t*)carve((size_t)MROWS * 512 * 2);
    bf16_t* hb       = (bf16_t*)carve((size_t)MROWS * 2048 * 2);
    (void)ws_size; (void)n_in; (void)in_sizes; (void)out_size;

    // 0) weights -> bf16
    auto cvt = [&](const float* s, bf16_t* d, int n) {
        cvt_f32_to_bf16<<<(n + 255) / 256, 256, 0, stream>>>(s, d, n);
    };
    cvt(inpw, w_in_bf,  1536 * 512);
    cvt(outw, w_out_bf, 512 * 512);
    cvt(w1,   w1_bf,    2048 * 512);
    cvt(w2,   w2_bf,    512 * 2048);

    // 1) LN1 + RoPE
    ln_kernel<true><<<MROWS, 256, 0, stream>>>(src, ln1w, ln1b, rcos, rsin, xln, xr);

    // 2) QK projection: [8192,512] x [1024,512]^T
    gemm_bf16<EPI_BF16><<<dim3(8, 64), 256, 0, stream>>>(
        xr, w_in_bf, inpb, nullptr, qkb, nullptr, MROWS, 1024, 512, 512);
    // 3) V projection (uses un-rotated x)
    gemm_bf16<EPI_BF16><<<dim3(4, 64), 256, 0, stream>>>(
        xln, w_in_bf + (size_t)1024 * 512, inpb + 1024, nullptr, vb, nullptr, MROWS, 512, 512, 512);

    // 4) causal flash attention
    attn_kernel<<<dim3(64, 16), 128, 0, stream>>>(qkb, vb, attnb);

    // 5) out-proj + residual (fp32): res1 = src + attn @ Wo^T + bo
    gemm_bf16<EPI_RES><<<dim3(4, 64), 256, 0, stream>>>(
        attnb, w_out_bf, outb, src, nullptr, res1, MROWS, 512, 512, 512);

    // 6) LN2
    ln_kernel<false><<<MROWS, 256, 0, stream>>>(res1, ln2w, ln2b, nullptr, nullptr, yb, nullptr);

    // 7) FFN1 + exact GELU
    gemm_bf16<EPI_GELU><<<dim3(16, 64), 256, 0, stream>>>(
        yb, w1_bf, b1, nullptr, hb, nullptr, MROWS, 2048, 512, 512);

    // 8) FFN2 + residual -> output (fp32)
    gemm_bf16<EPI_RES><<<dim3(4, 64), 256, 0, stream>>>(
        hb, w2_bf, b2, res1, nullptr, outp, MROWS, 512, 2048, 2048);
}
